// MaskedDeepDAN_60052232732819
// MI455X (gfx1250) — matrix operand order repaired and checked by reading, not compiled, not run on hardware
//
#include <hip/hip_runtime.h>
#include <stdint.h>

// ---------------------------------------------------------------------------
// MaskedDeepDAN on MI455X (gfx1250): bf16 WMMA (fp32 accumulate) fused layers,
// triple-buffered async global->LDS staging (ASYNCcnt) + ds_load_b128 frags.
// ---------------------------------------------------------------------------

typedef __attribute__((ext_vector_type(16))) __bf16 v16bf;
typedef __attribute__((ext_vector_type(8)))  float  v8f;

#define B_ROWS 8192
#define D_IN   2048
#define H_DIM  1024
#define C_OUT  1000

// LDS staging geometry: stage = A(128x32 bf16) + B(128x32 bf16),
// rows padded to 40 elements (80B) to spread ds_load_b128 bank groups.
#define ROW_ELEMS   40           // 32 data + 8 pad
#define TILE_ELEMS  (128 * ROW_ELEMS)       // 5120 elems = 10240 B
#define STAGE_ELEMS (2 * TILE_ELEMS)        // A + B     = 20480 B
#define STAGE_BYTES (STAGE_ELEMS * 2)
#define NBUF 3

union FragU {
  v16bf v;
  uint4 q[2];
};

__device__ __forceinline__ uint32_t f32_to_bf16(float f) {
  uint32_t u = __float_as_uint(f);
  uint32_t r = (u + 0x7FFFu + ((u >> 16) & 1u)) >> 16;   // round to nearest even
  return r & 0xFFFFu;
}

// ------------------------- conversion kernels ------------------------------

__global__ void cvt_f32_to_bf16_x4(const float* __restrict__ in,
                                   unsigned short* __restrict__ out, long n4) {
  long i = (long)blockIdx.x * blockDim.x + threadIdx.x;
  if (i >= n4) return;
  const float4 f = reinterpret_cast<const float4*>(in)[i];
  uint2 u;
  u.x = f32_to_bf16(f.x) | (f32_to_bf16(f.y) << 16);
  u.y = f32_to_bf16(f.z) | (f32_to_bf16(f.w) << 16);
  reinterpret_cast<uint2*>(out)[i] = u;
}

// mask-and-convert a weight matrix; rows >= `rows` (padding) are zeroed.
__global__ void mask_cvt_x4(const float* __restrict__ W,
                            const unsigned char* __restrict__ m,
                            unsigned short* __restrict__ out,
                            long rows, long cols, long padRows) {
  long i4 = ((long)blockIdx.x * blockDim.x + threadIdx.x) * 4;
  long total = padRows * cols;
  if (i4 >= total) return;
  long r = i4 / cols;               // cols % 4 == 0 -> same row for all 4
  uint2 u;
  if (r < rows) {
    const float4 f = *reinterpret_cast<const float4*>(W + i4);
    uchar4 mm = *reinterpret_cast<const uchar4*>(m + i4);
    uint32_t e0 = mm.x ? f32_to_bf16(f.x) : 0u;
    uint32_t e1 = mm.y ? f32_to_bf16(f.y) : 0u;
    uint32_t e2 = mm.z ? f32_to_bf16(f.z) : 0u;
    uint32_t e3 = mm.w ? f32_to_bf16(f.w) : 0u;
    u.x = e0 | (e1 << 16);
    u.y = e2 | (e3 << 16);
  } else {
    u.x = 0u; u.y = 0u;
  }
  *reinterpret_cast<uint2*>(out + i4) = u;
}

// ------------------------- fused multi-term GEMM ---------------------------
// out = sum_t relu_t( A_t[M,K] @ W_t[N,K]^T + bias_t )   (relu optional)
// Block 256 threads = 8 waves (4 M x 2 N); wave tile 32x64 = 2x4 16x16 WMMA
// frags; block tile 128x128. Stages of K=32 staged to LDS with async copies.

struct Terms {
  const unsigned short* A[4];
  const unsigned short* W[4];
  const float*          bias[4];
};

// Issue the async copies for one stage: 256 threads x (2 A + 2 B) x 16B.
__device__ __forceinline__ void issue_stage(const unsigned short* __restrict__ A,
                                            const unsigned short* __restrict__ W,
                                            int K, int mBlock, int nBlock,
                                            int k0, uint32_t ldsStageBase) {
#pragma unroll
  for (int h = 0; h < 2; ++h) {
    const int j   = (int)threadIdx.x + h * 256;
    const int row = j >> 2;
    const int c   = j & 3;                       // 16B chunk within the 64B row
    const unsigned long long gA =
        (unsigned long long)(uintptr_t)(A + (long)(mBlock + row) * K + k0 + c * 8);
    const uint32_t lA = ldsStageBase + (uint32_t)(row * 80 + c * 16);
    asm volatile("global_load_async_to_lds_b128 %0, %1, off"
                 :: "v"(lA), "v"(gA) : "memory");
    const unsigned long long gB =
        (unsigned long long)(uintptr_t)(W + (long)(nBlock + row) * K + k0 + c * 8);
    const uint32_t lB = ldsStageBase + 10240u + (uint32_t)(row * 80 + c * 16);
    asm volatile("global_load_async_to_lds_b128 %0, %1, off"
                 :: "v"(lB), "v"(gB) : "memory");
  }
}

__launch_bounds__(256)
__global__ void gemm_terms(Terms t, int nTerms, int K,
                           unsigned short* __restrict__ outBf16, // ld 1024 (or null)
                           float* __restrict__ outF32,           // ld outN  (or null)
                           int outN, int biasN, int applyRelu) {
  __shared__ __align__(16) unsigned short smem[NBUF * STAGE_ELEMS];  // 60 KB

  const int lane  = threadIdx.x & 31;
  const int wave  = threadIdx.x >> 5;
  const int waveM = wave & 3;
  const int waveN = wave >> 2;
  const int mBlock = blockIdx.y * 128;
  const int nBlock = blockIdx.x * 128;
  const int lmod  = lane & 15;
  const int lhi   = lane >> 4;     // 0 or 1
  const int aOff  = lhi * 8;       // A frag: K in {0..7,16..23} / {8..15,24..31}
  const int bOff  = lhi * 8;       // B frag: K in {0..7,16..23} / {8..15,24..31}

  const uint32_t ldsBase = (uint32_t)(uintptr_t)&smem[0];

  // per-wave LDS element bases for fragment reads (row-padded layout)
  int aRowLocal[2], bRowLocal[4];
#pragma unroll
  for (int mf = 0; mf < 2; ++mf)
    aRowLocal[mf] = (waveM * 32 + mf * 16 + lmod) * ROW_ELEMS;
#pragma unroll
  for (int nf = 0; nf < 4; ++nf)
    bRowLocal[nf] = (waveN * 64 + nf * 16 + lmod) * ROW_ELEMS;

  const v8f zero8 = {0.f, 0.f, 0.f, 0.f, 0.f, 0.f, 0.f, 0.f};
  v8f tot[2][4];
#pragma unroll
  for (int mf = 0; mf < 2; ++mf)
#pragma unroll
    for (int nf = 0; nf < 4; ++nf) tot[mf][nf] = zero8;

  for (int tI = 0; tI < nTerms; ++tI) {
    const unsigned short* __restrict__ A = t.A[tI];
    const unsigned short* __restrict__ W = t.W[tI];
    const int nStages = K >> 5;

    v8f acc[2][4];
#pragma unroll
    for (int mf = 0; mf < 2; ++mf)
#pragma unroll
      for (int nf = 0; nf < 4; ++nf) acc[mf][nf] = zero8;

    __syncthreads();  // previous term (or epilogue-free entry) done with buffers
    issue_stage(A, W, K, mBlock, nBlock, 0, ldsBase);
    issue_stage(A, W, K, mBlock, nBlock, 32, ldsBase + STAGE_BYTES);

    int sel = 0;        // stage % NBUF
    int sel2 = 2;       // (stage+2) % NBUF
    for (int s = 0; s < nStages; ++s) {
      // my 4 copies for stage s are the oldest outstanding; done-in-order =>
      // waiting <=4 (next stage still in flight) proves stage s landed.
      if (s + 1 < nStages)
        asm volatile("s_wait_asynccnt 0x4" ::: "memory");
      else
        asm volatile("s_wait_asynccnt 0x0" ::: "memory");
      __syncthreads();  // whole stage s visible; everyone done reading s-1

      const unsigned short* sA = &smem[sel * STAGE_ELEMS];
      const unsigned short* sB = sA + TILE_ELEMS;
      FragU a[2], b[4];
#pragma unroll
      for (int mf = 0; mf < 2; ++mf) {
        const unsigned short* p = sA + aRowLocal[mf];
        a[mf].q[0] = *reinterpret_cast<const uint4*>(p + aOff);
        a[mf].q[1] = *reinterpret_cast<const uint4*>(p + 16 + aOff);
      }
#pragma unroll
      for (int nf = 0; nf < 4; ++nf) {
        const unsigned short* p = sB + bRowLocal[nf];
        b[nf].q[0] = *reinterpret_cast<const uint4*>(p + bOff);
        b[nf].q[1] = *reinterpret_cast<const uint4*>(p + bOff + 16);
      }

      // refill the buffer stage s-1 used (all reads of it completed before the
      // barrier above); targets neither stage s nor s+1.
      if (s + 2 < nStages)
        issue_stage(A, W, K, mBlock, nBlock, (s + 2) << 5,
                    ldsBase + (uint32_t)sel2 * STAGE_BYTES);

#pragma unroll
      for (int mf = 0; mf < 2; ++mf)
#pragma unroll
        for (int nf = 0; nf < 4; ++nf)
          acc[mf][nf] = __builtin_amdgcn_wmma_f32_16x16x32_bf16(
              false, a[mf].v, false, b[nf].v, (short)0, acc[mf][nf],
              false, false);

      sel = (sel == NBUF - 1) ? 0 : sel + 1;
      sel2 = (sel2 == NBUF - 1) ? 0 : sel2 + 1;
    }

    const float* __restrict__ bias = t.bias[tI];
#pragma unroll
    for (int nf = 0; nf < 4; ++nf) {
      const int n = nBlock + waveN * 64 + nf * 16 + lmod;
      const float bv = (n < biasN) ? bias[n] : 0.f;
#pragma unroll
      for (int mf = 0; mf < 2; ++mf)
#pragma unroll
        for (int r = 0; r < 8; ++r) {
          float v = acc[mf][nf][r] + bv;
          if (applyRelu) v = fmaxf(v, 0.f);
          tot[mf][nf][r] += v;
        }
    }
  }

  // store: C/D layout -> VGPR r, lane<16: M=r, lane>=16: M=r+8; N = lane%16
#pragma unroll
  for (int nf = 0; nf < 4; ++nf) {
    const int n = nBlock + waveN * 64 + nf * 16 + lmod;
#pragma unroll
    for (int mf = 0; mf < 2; ++mf) {
      const int mBase = mBlock + waveM * 32 + mf * 16 + lhi * 8;
#pragma unroll
      for (int r = 0; r < 8; ++r) {
        const float v = tot[mf][nf][r];
        const long row = mBase + r;
        if (outBf16)
          outBf16[row * (long)H_DIM + n] = (unsigned short)f32_to_bf16(v);
        if (outF32 && n < outN)
          outF32[row * (long)outN + n] = v;
      }
    }
  }
}

// ------------------------------ host side ----------------------------------

extern "C" void kernel_launch(void* const* d_in, const int* in_sizes, int n_in,
                              void* d_out, int out_size, void* d_ws, size_t ws_size,
                              hipStream_t stream) {
  (void)in_sizes; (void)n_in; (void)out_size; (void)ws_size;

  // setup_inputs dict order (lists flattened):
  const float*         x   = (const float*)d_in[0];
  const float*         Wf  = (const float*)d_in[1];
  const float*         bf  = (const float*)d_in[2];
  const unsigned char* mf  = (const unsigned char*)d_in[3];
  const float*         mainW[4]; const float* mainB[4]; const unsigned char* mainM[4];
  for (int i = 0; i < 4; ++i) {
    mainW[i] = (const float*)d_in[4 + i];
    mainB[i] = (const float*)d_in[8 + i];
    mainM[i] = (const unsigned char*)d_in[12 + i];
  }
  const float* skipW[6]; const float* skipB[6]; const unsigned char* skipM[6];
  for (int k = 0; k < 6; ++k) {
    skipW[k] = (const float*)d_in[16 + k];
    skipB[k] = (const float*)d_in[22 + k];
    skipM[k] = (const unsigned char*)d_in[28 + k];
  }
  const float*         Wo  = (const float*)d_in[34];
  const float*         bo  = (const float*)d_in[35];
  const unsigned char* mo  = (const unsigned char*)d_in[36];
  float*               out = (float*)d_out;

  // workspace carve-up (all bf16 = ushort buffers)
  char*  ws  = (char*)d_ws;
  size_t off = 0;
  auto carve = [&](long elems) -> unsigned short* {
    unsigned short* p = (unsigned short*)(ws + off);
    off = (off + (size_t)elems * 2 + 255) & ~(size_t)255;
    return p;
  };
  unsigned short* xB   = carve((long)B_ROWS * D_IN);         // 32 MB
  unsigned short* WfM  = carve((long)H_DIM * D_IN);          //  4 MB
  unsigned short* WmM[4]; for (int i = 0; i < 4; ++i) WmM[i] = carve((long)H_DIM * H_DIM);
  unsigned short* WsM[6]; for (int k = 0; k < 6; ++k) WsM[k] = carve((long)H_DIM * H_DIM);
  unsigned short* WoM  = carve((long)H_DIM * H_DIM);         // padded 1024x1024
  unsigned short* res[5]; for (int i = 0; i < 5; ++i) res[i] = carve((long)B_ROWS * H_DIM);

  const int TPB = 256;

  // 1) convert x to bf16
  {
    long n4 = (long)B_ROWS * D_IN / 4;
    cvt_f32_to_bf16_x4<<<dim3((unsigned)((n4 + TPB - 1) / TPB)), dim3(TPB), 0, stream>>>(x, xB, n4);
  }
  // 2) mask+convert all weights
  auto mask_launch = [&](const float* W, const unsigned char* m, unsigned short* o,
                         long rows, long cols, long padRows) {
    long n4 = padRows * cols / 4;
    mask_cvt_x4<<<dim3((unsigned)((n4 + TPB - 1) / TPB)), dim3(TPB), 0, stream>>>(W, m, o, rows, cols, padRows);
  };
  mask_launch(Wf, mf, WfM, H_DIM, D_IN, H_DIM);
  for (int i = 0; i < 4; ++i) mask_launch(mainW[i], mainM[i], WmM[i], H_DIM, H_DIM, H_DIM);
  for (int k = 0; k < 6; ++k) mask_launch(skipW[k], skipM[k], WsM[k], H_DIM, H_DIM, H_DIM);
  mask_launch(Wo, mo, WoM, C_OUT, H_DIM, H_DIM);  // zero-pad rows 1000..1023

  // 3) fused GEMM layers
  const dim3 gGrid(H_DIM / 128, B_ROWS / 128);  // (8, 64)
  const dim3 gBlk(TPB);
  auto gemm = [&](Terms& t, int nTerms, int K, unsigned short* oB, float* oF,
                  int outN, int biasN, int relu) {
    gemm_terms<<<gGrid, gBlk, 0, stream>>>(t, nTerms, K, oB, oF, outN, biasN, relu);
  };

  // layer 0: relu(x @ WfM^T + bf) -> res[0]
  { Terms t{}; t.A[0] = xB; t.W[0] = WfM; t.bias[0] = bf;
    gemm(t, 1, D_IN, res[0], nullptr, H_DIM, H_DIM, 1); }
  // layer 1: relu(res0 @ main0)
  { Terms t{}; t.A[0] = res[0]; t.W[0] = WmM[0]; t.bias[0] = mainB[0];
    gemm(t, 1, H_DIM, res[1], nullptr, H_DIM, H_DIM, 1); }
  // layer 2: relu(res1 @ main1) + relu(res0 @ skip0)
  { Terms t{};
    t.A[0] = res[1]; t.W[0] = WmM[1]; t.bias[0] = mainB[1];
    t.A[1] = res[0]; t.W[1] = WsM[0]; t.bias[1] = skipB[0];
    gemm(t, 2, H_DIM, res[2], nullptr, H_DIM, H_DIM, 1); }
  // layer 3: relu(res2 @ main2) + relu(res0 @ skip1) + relu(res1 @ skip2)
  { Terms t{};
    t.A[0] = res[2]; t.W[0] = WmM[2]; t.bias[0] = mainB[2];
    t.A[1] = res[0]; t.W[1] = WsM[1]; t.bias[1] = skipB[1];
    t.A[2] = res[1]; t.W[2] = WsM[2]; t.bias[2] = skipB[2];
    gemm(t, 3, H_DIM, res[3], nullptr, H_DIM, H_DIM, 1); }
  // layer 4: relu(res3 @ main3) + relu(res0 @ skip3) + relu(res1 @ skip4) + relu(res2 @ skip5)
  { Terms t{};
    t.A[0] = res[3]; t.W[0] = WmM[3]; t.bias[0] = mainB[3];
    t.A[1] = res[0]; t.W[1] = WsM[3]; t.bias[1] = skipB[3];
    t.A[2] = res[1]; t.W[2] = WsM[4]; t.bias[2] = skipB[4];
    t.A[3] = res[2]; t.W[3] = WsM[5]; t.bias[3] = skipB[5];
    gemm(t, 4, H_DIM, res[4], nullptr, H_DIM, H_DIM, 1); }
  // output: res4 @ WoM^T + bo (no relu), f32 -> d_out [8192,1000]
  { Terms t{}; t.A[0] = res[4]; t.W[0] = WoM; t.bias[0] = bo;
    gemm(t, 1, H_DIM, nullptr, out, C_OUT, C_OUT, 0); }
}
